// MultiHeadAttention_4398046511380
// MI455X (gfx1250) — matrix operand order repaired and checked by reading, not compiled, not run on hardware
//
#include <hip/hip_runtime.h>

// ---------------------------------------------------------------------------
// MI455X (gfx1250) fused MHA: bf16 WMMA GEMM projections + flash attention.
// B=2, S=2048, D=2048, H=16, DEPTH=128.
// Async global->LDS staging (ASYNCcnt path) for raw bf16 tile copies.
// ---------------------------------------------------------------------------

typedef __bf16 bf16;
typedef __attribute__((ext_vector_type(16))) bf16  v16bf;
typedef __attribute__((ext_vector_type(8)))  bf16  v8bf;
typedef __attribute__((ext_vector_type(4)))  bf16  v4bf;
typedef __attribute__((ext_vector_type(8)))  float v8f;
typedef __attribute__((ext_vector_type(4)))  float v4f;

#define WMMA_BF16(a, b, c) \
  __builtin_amdgcn_wmma_f32_16x16x32_bf16(false, (a), false, (b), (short)0, (c), false, false)

static __device__ inline v16bf cat16(v8bf lo, v8bf hi) {
  v16bf r;
#pragma unroll
  for (int i = 0; i < 8; ++i) { r[i] = lo[i]; r[i + 8] = hi[i]; }
  return r;
}
static __device__ inline v8f zero8() {
  v8f r;
#pragma unroll
  for (int i = 0; i < 8; ++i) r[i] = 0.0f;
  return r;
}

// ---- async global->LDS 16-byte copy (CDNA5 GLOBAL_LOAD_ASYNC_TO_LDS_B128) --
// Builtin signature (from hipcc diagnostic): param0 = int32x4 in AS(1) (global),
// param1 = LDS side, then imm offset + imm cpol.
typedef int i32x4 __attribute__((vector_size(16)));
typedef __attribute__((address_space(1))) i32x4 as1_i32x4;
typedef __attribute__((address_space(3))) i32x4 as3_i32x4;

static __device__ inline void async_copy16(const bf16* g, bf16* l) {
#if __has_builtin(__builtin_amdgcn_global_load_async_to_lds_b128)
  __builtin_amdgcn_global_load_async_to_lds_b128((as1_i32x4*)g, (as3_i32x4*)l, 0, 0);
#else
  asm volatile("global_load_async_to_lds_b128 %0, %1, off"
               :: "v"((unsigned int)(unsigned long long)l), "v"(g)
               : "memory");
#endif
}
static __device__ inline void async_wait_all() {
#if __has_builtin(__builtin_amdgcn_s_wait_asynccnt)
  __builtin_amdgcn_s_wait_asynccnt(0);
#else
  asm volatile("s_wait_asynccnt 0" ::: "memory");
#endif
}

constexpr int S_LEN = 2048;
constexpr int D_DIM = 2048;
constexpr int HEADS = 16;
constexpr int DEPTH = 128;
constexpr int LDS_AW = 40;   // 32-wide bf16 tile stride (+8 pad, keeps 16B align)

// ---------------------------------------------------------------------------
// Projection GEMM: out = relu(A[4096x2048] @ W[2048x2048] + bias), bf16 out.
// MODE 0: out[((b*H+h)*S + s)*128 + d]   (Q, K: per-head row major)
// MODE 1: out[((b*H+h)*128 + d)*S + s]   (V: per-head transposed)
// Block tile 128x128, 8 waves, wave tile 64x32 (4x2 WMMA 16x16 accumulators).
// fp32 sources need on-the-fly bf16 conversion -> VALU staging path kept.
// ---------------------------------------------------------------------------
template <int MODE>
__global__ __launch_bounds__(256) void proj_gemm(const float* __restrict__ A,
                                                 const float* __restrict__ W,
                                                 const float* __restrict__ bias,
                                                 bf16* __restrict__ out) {
  __shared__ bf16 sA[128 * LDS_AW];
  __shared__ bf16 sW[128 * LDS_AW];   // stored transposed: [ncol][k]

  const int t    = threadIdx.x;
  const int lane = t & 31;
  const int wid  = t >> 5;
  const int wm   = wid >> 2;     // 0..1 : 64 rows
  const int wn   = wid & 3;      // 0..3 : 32 cols
  const int l16  = lane & 15;
  const int lh   = lane >> 4;
  const int bn   = blockIdx.x;   // N/128
  const int bm   = blockIdx.y;   // M/128

  v8f acc[4][2];
#pragma unroll
  for (int i = 0; i < 4; ++i)
#pragma unroll
    for (int j = 0; j < 2; ++j) acc[i][j] = zero8();

  const int arow  = t >> 1;          // 0..127
  const int acol0 = (t & 1) * 16;    // 0 / 16
  const int wk    = t >> 3;          // 0..31
  const int wc0   = (t & 7) * 16;    // 0..112

  for (int kk = 0; kk < D_DIM; kk += 32) {
    __syncthreads();
    // ---- stage A tile (fp32 -> bf16), 128x32
    {
      const float* ap = A + (long)(bm * 128 + arow) * D_DIM + kk + acol0;
#pragma unroll
      for (int i = 0; i < 4; ++i) {
        v4f f = *(const v4f*)(ap + i * 4);
        v4bf pk;
#pragma unroll
        for (int j = 0; j < 4; ++j) pk[j] = (bf16)f[j];
        *(v4bf*)&sA[arow * LDS_AW + acol0 + i * 4] = pk;
      }
    }
    // ---- stage W tile transposed (fp32 -> bf16), sW[col][k], 128x32
    {
      const float* wp = W + (long)(kk + wk) * D_DIM + bn * 128 + wc0;
#pragma unroll
      for (int i = 0; i < 4; ++i) {
        v4f f = *(const v4f*)(wp + i * 4);
#pragma unroll
        for (int j = 0; j < 4; ++j)
          sW[(wc0 + i * 4 + j) * LDS_AW + wk] = (bf16)f[j];
      }
    }
    __syncthreads();

    // ---- fragments + WMMA
    v16bf af[4];
#pragma unroll
    for (int i = 0; i < 4; ++i) {
      const bf16* p = &sA[(wm * 64 + i * 16 + l16) * LDS_AW + lh * 8];
      af[i] = cat16(*(const v8bf*)p, *(const v8bf*)(p + 16));
    }
    v16bf bfr[2];
#pragma unroll
    for (int j = 0; j < 2; ++j) {
      const bf16* p = &sW[(wn * 32 + j * 16 + l16) * LDS_AW + lh * 8];
      bfr[j] = cat16(*(const v8bf*)p, *(const v8bf*)(p + 16));
    }
#pragma unroll
    for (int i = 0; i < 4; ++i)
#pragma unroll
      for (int j = 0; j < 2; ++j) acc[i][j] = WMMA_BF16(af[i], bfr[j], acc[i][j]);
  }

  // ---- epilogue: bias + relu + scatter to head layout
#pragma unroll
  for (int j = 0; j < 2; ++j) {
    const int coll = bn * 128 + wn * 32 + j * 16 + l16;
    const float bia = bias[coll];
    const int h = coll >> 7, d = coll & 127;
#pragma unroll
    for (int i = 0; i < 4; ++i) {
#pragma unroll
      for (int g = 0; g < 8; ++g) {
        const int m = bm * 128 + wm * 64 + i * 16 + g + (lh ? 8 : 0);
        float v = acc[i][j][g] + bia;
        v = v > 0.0f ? v : 0.0f;
        const int bb = m >> 11, s = m & 2047;
        long idx;
        if (MODE == 0) idx = (((long)(bb * HEADS + h)) * S_LEN + s) * DEPTH + d;
        else           idx = (((long)(bb * HEADS + h)) * DEPTH + d) * S_LEN + s;
        out[idx] = (bf16)v;
      }
    }
  }
}

// ---------------------------------------------------------------------------
// Flash attention, causal, bf16 WMMA, f32 online softmax.
// grid = (S/64, B*H), block = 128 (4 waves); wave owns 16 q rows x 128 depth.
// q,k: [B*H][S][128]   v: [B*H][128][S]   o: [B][S][D] (concat heads), bf16.
// K/V tiles staged with async global->LDS copies (ASYNCcnt).
// ---------------------------------------------------------------------------
__global__ __launch_bounds__(128) void flash_attn(const bf16* __restrict__ q,
                                                  const bf16* __restrict__ k,
                                                  const bf16* __restrict__ v,
                                                  bf16* __restrict__ o) {
  __shared__ bf16 sK[64 * 136];       // [key][depth], +8 pad
  __shared__ bf16 sV[128 * 72];       // [depth][key], +8 pad
  __shared__ bf16 sP[4 * 16 * 72];    // per-wave P scratch [row][key]

  const int t   = threadIdx.x;
  const int lane = t & 31, wid = t >> 5;
  const int l16 = lane & 15, lh = lane >> 4;
  const int qb  = blockIdx.x;     // q tile of 64 rows
  const int bh  = blockIdx.y;     // b*16+h
  const long base = (long)bh * S_LEN * DEPTH;
  const int q0 = qb * 64 + wid * 16;

  // Q fragments for this wave's 16 rows (direct from global, A layout)
  v16bf qf[4];
  {
    const bf16* qp = q + base + (long)(q0 + l16) * DEPTH;
#pragma unroll
    for (int kc = 0; kc < 4; ++kc) {
      const bf16* p = qp + kc * 32 + lh * 8;
      qf[kc] = cat16(*(const v8bf*)p, *(const v8bf*)(p + 16));
    }
  }

  v8f oacc[8];
#pragma unroll
  for (int dt = 0; dt < 8; ++dt) oacc[dt] = zero8();
  float mrun[8], lrun[8];
#pragma unroll
  for (int g = 0; g < 8; ++g) { mrun[g] = -3.0e38f; lrun[g] = 0.0f; }

  const float kscale = 0.08838834764831845f * 1.4426950408889634f; // 1/sqrt(128)*log2e
  bf16* pw = &sP[wid * 16 * 72];

  for (int kt = 0; kt <= qb; ++kt) {
    __syncthreads();
    // ---- stage K tile [64][128] and V^T tile [128][64] via async LDS DMA
    {
      const int row = t >> 1, half = t & 1;
      const bf16* kp = k + base + (long)(kt * 64 + row) * DEPTH + half * 64;
      bf16* kd = &sK[row * 136 + half * 64];
#pragma unroll
      for (int i = 0; i < 8; ++i) async_copy16(kp + i * 8, kd + i * 8);
      const bf16* vp = v + base + (long)t * S_LEN + kt * 64;
      bf16* vd = &sV[t * 72];
#pragma unroll
      for (int i = 0; i < 8; ++i) async_copy16(vp + i * 8, vd + i * 8);
    }
    if (kt < qb) {  // prefetch next tile into caches (global_prefetch_b8)
      __builtin_prefetch(k + base + (long)((kt + 1) * 64 + (t >> 1)) * DEPTH, 0, 1);
      __builtin_prefetch(v + base + (long)t * S_LEN + (kt + 1) * 64, 0, 1);
    }
    async_wait_all();
    __syncthreads();

    // ---- scores: S = Q (16x128) @ K^T (128x64), 4 n-tiles of 16 keys
    v8f sc[4];
#pragma unroll
    for (int nt = 0; nt < 4; ++nt) sc[nt] = zero8();
#pragma unroll
    for (int kc = 0; kc < 4; ++kc) {
#pragma unroll
      for (int nt = 0; nt < 4; ++nt) {
        const bf16* p = &sK[(nt * 16 + l16) * 136 + kc * 32 + lh * 8];
        v16bf bm = cat16(*(const v8bf*)p, *(const v8bf*)(p + 16));
        sc[nt] = WMMA_BF16(qf[kc], bm, sc[nt]);
      }
    }

    // ---- causal mask + online softmax (base-2 domain)
#pragma unroll
    for (int g = 0; g < 8; ++g) {
      const int qr = q0 + g + (lh ? 8 : 0);
      float mx = -3.0e38f;
#pragma unroll
      for (int nt = 0; nt < 4; ++nt) {
        const int key = kt * 64 + nt * 16 + l16;
        float sv = (key <= qr) ? sc[nt][g] * kscale : -3.0e38f;
        sc[nt][g] = sv;
        mx = fmaxf(mx, sv);
      }
#pragma unroll
      for (int off = 1; off < 16; off <<= 1) mx = fmaxf(mx, __shfl_xor(mx, off, 32));
      const float mnew = fmaxf(mrun[g], mx);
      const float c = exp2f(mrun[g] - mnew);
      mrun[g] = mnew;
      float rsum = 0.0f;
#pragma unroll
      for (int nt = 0; nt < 4; ++nt) {
        const float pv = exp2f(sc[nt][g] - mnew);
        sc[nt][g] = pv;
        rsum += pv;
      }
#pragma unroll
      for (int off = 1; off < 16; off <<= 1) rsum += __shfl_xor(rsum, off, 32);
      lrun[g] = lrun[g] * c + rsum;
#pragma unroll
      for (int dt = 0; dt < 8; ++dt) oacc[dt][g] *= c;
      // spill P row to per-wave LDS scratch (C layout -> A layout round trip)
      const int r = g + (lh ? 8 : 0);
#pragma unroll
      for (int nt = 0; nt < 4; ++nt) pw[r * 72 + nt * 16 + l16] = (bf16)sc[nt][g];
    }
    asm volatile("s_wait_dscnt 0" ::: "memory");  // LDS RAW inside the wave

    // ---- O += P (16x64) @ V (64x128)
    v16bf pf[2];
#pragma unroll
    for (int ks = 0; ks < 2; ++ks) {
      const bf16* p = &pw[l16 * 72 + ks * 32 + lh * 8];
      pf[ks] = cat16(*(const v8bf*)p, *(const v8bf*)(p + 16));
    }
#pragma unroll
    for (int dt = 0; dt < 8; ++dt) {
#pragma unroll
      for (int ks = 0; ks < 2; ++ks) {
        const bf16* p = &sV[(dt * 16 + l16) * 72 + ks * 32 + lh * 8];
        v16bf bm = cat16(*(const v8bf*)p, *(const v8bf*)(p + 16));
        oacc[dt] = WMMA_BF16(pf[ks], bm, oacc[dt]);
      }
    }
  }

  // ---- epilogue: normalize, write concat-head bf16 [B][S][D]
  const int h = bh & 15, bb = bh >> 4;
#pragma unroll
  for (int g = 0; g < 8; ++g) {
    const float linv = 1.0f / lrun[g];
    const int s = q0 + g + (lh ? 8 : 0);
    bf16* op = o + ((long)(bb * S_LEN + s)) * D_DIM + h * DEPTH;
#pragma unroll
    for (int dt = 0; dt < 8; ++dt) op[dt * 16 + l16] = (bf16)(oacc[dt][g] * linv);
  }
}

// ---------------------------------------------------------------------------
// Output GEMM: out = relu(Abf16[4096x2048] @ Wo + bo), fp32 out [B][S][D].
// A tile is a raw bf16 copy -> async global->LDS staging.
// ---------------------------------------------------------------------------
__global__ __launch_bounds__(256) void out_gemm(const bf16* __restrict__ A,
                                                const float* __restrict__ W,
                                                const float* __restrict__ bias,
                                                float* __restrict__ out) {
  __shared__ bf16 sA[128 * LDS_AW];
  __shared__ bf16 sW[128 * LDS_AW];

  const int t = threadIdx.x;
  const int lane = t & 31, wid = t >> 5;
  const int wm = wid >> 2, wn = wid & 3;
  const int l16 = lane & 15, lh = lane >> 4;
  const int bn = blockIdx.x, bm = blockIdx.y;

  v8f acc[4][2];
#pragma unroll
  for (int i = 0; i < 4; ++i)
#pragma unroll
    for (int j = 0; j < 2; ++j) acc[i][j] = zero8();

  const int arow = t >> 1;
  const int acol0 = (t & 1) * 16;
  const int wk = t >> 3;
  const int wc0 = (t & 7) * 16;

  for (int kk = 0; kk < D_DIM; kk += 32) {
    __syncthreads();
    {
      const bf16* ap = A + (long)(bm * 128 + arow) * D_DIM + kk + acol0;
      bf16* dst = &sA[arow * LDS_AW + acol0];
      async_copy16(ap, dst);
      async_copy16(ap + 8, dst + 8);
    }
    {
      const float* wp = W + (long)(kk + wk) * D_DIM + bn * 128 + wc0;
#pragma unroll
      for (int i = 0; i < 4; ++i) {
        v4f f = *(const v4f*)(wp + i * 4);
#pragma unroll
        for (int j = 0; j < 4; ++j)
          sW[(wc0 + i * 4 + j) * LDS_AW + wk] = (bf16)f[j];
      }
    }
    async_wait_all();
    __syncthreads();

    v16bf af[4];
#pragma unroll
    for (int i = 0; i < 4; ++i) {
      const bf16* p = &sA[(wm * 64 + i * 16 + l16) * LDS_AW + lh * 8];
      af[i] = cat16(*(const v8bf*)p, *(const v8bf*)(p + 16));
    }
    v16bf bfr[2];
#pragma unroll
    for (int j = 0; j < 2; ++j) {
      const bf16* p = &sW[(wn * 32 + j * 16 + l16) * LDS_AW + lh * 8];
      bfr[j] = cat16(*(const v8bf*)p, *(const v8bf*)(p + 16));
    }
#pragma unroll
    for (int i = 0; i < 4; ++i)
#pragma unroll
      for (int j = 0; j < 2; ++j) acc[i][j] = WMMA_BF16(af[i], bfr[j], acc[i][j]);
  }

#pragma unroll
  for (int j = 0; j < 2; ++j) {
    const int coll = bn * 128 + wn * 32 + j * 16 + l16;
    const float bia = bias[coll];
#pragma unroll
    for (int i = 0; i < 4; ++i) {
#pragma unroll
      for (int g = 0; g < 8; ++g) {
        const int m = bm * 128 + wm * 64 + i * 16 + g + (lh ? 8 : 0);
        float vv = acc[i][j][g] + bia;
        out[(long)m * D_DIM + coll] = vv > 0.0f ? vv : 0.0f;
      }
    }
  }
}

// ---------------------------------------------------------------------------
extern "C" void kernel_launch(void* const* d_in, const int* in_sizes, int n_in,
                              void* d_out, int out_size, void* d_ws, size_t ws_size,
                              hipStream_t stream) {
  (void)in_sizes; (void)n_in; (void)out_size; (void)ws_size;
  const float* Q    = (const float*)d_in[0];
  const float* K    = (const float*)d_in[1];
  const float* V    = (const float*)d_in[2];
  // d_in[3] = mask: causal, computed from indices in-kernel (not read)
  const float* Wq = (const float*)d_in[4];
  const float* bq = (const float*)d_in[5];
  const float* Wk = (const float*)d_in[6];
  const float* bk = (const float*)d_in[7];
  const float* Wv = (const float*)d_in[8];
  const float* bv = (const float*)d_in[9];
  const float* Wo = (const float*)d_in[10];
  const float* bo = (const float*)d_in[11];

  const long SZE = (long)2 * HEADS * S_LEN * DEPTH;  // 8,388,608 bf16 elements
  bf16* qws = (bf16*)d_ws;
  bf16* kws = qws + SZE;
  bf16* vws = qws + 2 * SZE;
  bf16* aws = qws + 3 * SZE;

  dim3 gp(D_DIM / 128, (2 * S_LEN) / 128);  // (16, 32)
  proj_gemm<0><<<gp, 256, 0, stream>>>(Q, Wq, bq, qws);
  proj_gemm<0><<<gp, 256, 0, stream>>>(K, Wk, bk, kws);
  proj_gemm<1><<<gp, 256, 0, stream>>>(V, Wv, bv, vws);

  flash_attn<<<dim3(S_LEN / 64, 2 * HEADS), 128, 0, stream>>>(qws, kws, vws, aws);

  out_gemm<<<gp, 256, 0, stream>>>(aws, Wo, bo, (float*)d_out);
}
